// PairWeightedAveraging_58128087384617
// MI455X (gfx1250) — matrix operand order repaired and checked by reading, not compiled, not run on hardware
//
#include <hip/hip_runtime.h>
#include <hip/hip_bf16.h>

// ---------------------------------------------------------------------------
// PairWeightedAveraging on MI455X (gfx1250, wave32, WMMA)
//   S=512, T=512, CM=64, CZ=128, H=8, D=32, ND=H*D=256
// All GEMMs: bf16 operands / fp32 accumulate via v_wmma_f32_16x16x32_bf16.
// Fragments are loaded as contiguous 16-byte chunks (ds_load_b128 /
// global_load_b128) by pre-swizzling B operands into fragment-major order.
// ---------------------------------------------------------------------------

#define Sdim 512
#define Tdim 512
#define CMd  64
#define CZd  128
#define Hd   8
#define Dd   32
#define NDd  256             // H*D
#define ROWS_M (Sdim*Tdim)   // 262144 rows of m / out
#define ROWS_Z (Tdim*Tdim)   // 262144 rows of z

typedef __attribute__((ext_vector_type(16))) __bf16         v16bf;
typedef __attribute__((ext_vector_type(8)))  float          v8f;
typedef __attribute__((ext_vector_type(8)))  unsigned short us8;   // 16 bytes

union BF16Frag { v16bf v; us8 h[2]; unsigned short u[16]; };

// fp32 -> bf16 bits (round to nearest even)
static __device__ __forceinline__ unsigned short f2bfbits(float f) {
    union { float f; unsigned u; } c; c.f = f;
    unsigned r = c.u + 0x7FFFu + ((c.u >> 16) & 1u);
    return (unsigned short)(r >> 16);
}
// bf16 bits -> fp32
static __device__ __forceinline__ float bf2f(unsigned u16bits) {
    union { unsigned u; float f; } c; c.u = (u16bits & 0xFFFFu) << 16;
    return c.f;
}
static __device__ __forceinline__ unsigned packbf(float a, float b) {
    return (unsigned)f2bfbits(a) | ((unsigned)f2bfbits(b) << 16);
}
// single v_rcp_f32 (results are rounded to bf16 anyway)
static __device__ __forceinline__ float fastrcp(float x) {
    return __builtin_amdgcn_rcpf(x);
}

// ---------------------------------------------------------------------------
// Kernel 0: convert + swizzle weights into WMMA B-fragment-major order:
//   dst[((nt*KS + ks)*32 + lane)*16 + e]  holds  src[k = ks*32 + kr, n = nt*16 + (lane&15)]
//   with lane = (n&15) + (kr&16), e = kr&15  (B layout: lane=col, 16 contiguous K)
// ---------------------------------------------------------------------------
__global__ void k_cvt_w(const float* __restrict__ Wm, const float* __restrict__ Wg,
                        const float* __restrict__ Wo,
                        unsigned short* __restrict__ wmB, unsigned short* __restrict__ wgB,
                        unsigned short* __restrict__ woB) {
    int idx = blockIdx.x * 256 + threadIdx.x;
    if (idx >= CMd * NDd) return;
    // Wm/Wg: [K=64][N=256] -> KS=2, NT=16
    {
        int k = idx >> 8, n = idx & 255;
        int ks = k >> 5, kr = k & 31;
        int lane = (n & 15) + (kr & 16);
        int dst = (((n >> 4) * 2 + ks) * 32 + lane) * 16 + (kr & 15);
        wmB[dst] = f2bfbits(Wm[idx]);
        wgB[dst] = f2bfbits(Wg[idx]);
    }
    // Wo: [K=256][N=64] -> KS=8, NT=4
    {
        int k = idx >> 6, n = idx & 63;
        int ks = k >> 5, kr = k & 31;
        int lane = (n & 15) + (kr & 16);
        int dst = (((n >> 4) * 8 + ks) * 32 + lane) * 16 + (kr & 15);
        woB[dst] = f2bfbits(Wo[idx]);
    }
}

// ---------------------------------------------------------------------------
// Kernel 1: LayerNorm(m) then v = mn@Wm (bf16), g = sigmoid(mn@Wg) (bf16)
//   16 rows/block, 256 threads (8 waves). Each wave: 2 N-tiles of 16,
//   K=64 as 2 WMMA K-steps, for both Wm and Wg. B from swizzled weights.
// ---------------------------------------------------------------------------
__global__ void k_ln_m_project(const float* __restrict__ m,
                               const float* __restrict__ nw, const float* __restrict__ nb,
                               const unsigned short* __restrict__ WmB,
                               const unsigned short* __restrict__ WgB,
                               unsigned short* __restrict__ vbuf,
                               unsigned short* __restrict__ gbuf) {
    __shared__ unsigned short As[16][CMd];   // normalized rows, bf16
    const int tid = threadIdx.x;
    const long r0 = (long)blockIdx.x * 16;

    // --- stage 1: layernorm, 16 threads per row, 4 elems/thread ---
    {
        const int row = tid >> 4;
        const int c16 = tid & 15;
        const float* mr = m + (r0 + row) * CMd;
        float x[4]; float s = 0.f;
        #pragma unroll
        for (int q = 0; q < 4; ++q) { x[q] = mr[c16 * 4 + q]; s += x[q]; }
        #pragma unroll
        for (int off = 8; off >= 1; off >>= 1) s += __shfl_xor(s, off, 16);
        const float mu = s * (1.0f / CMd);
        float vs = 0.f;
        #pragma unroll
        for (int q = 0; q < 4; ++q) { float d = x[q] - mu; vs += d * d; }
        #pragma unroll
        for (int off = 8; off >= 1; off >>= 1) vs += __shfl_xor(vs, off, 16);
        const float rs = rsqrtf(vs * (1.0f / CMd) + 1e-5f);
        float y[4];
        #pragma unroll
        for (int q = 0; q < 4; ++q) {
            int c = c16 * 4 + q;
            y[q] = (x[q] - mu) * rs * nw[c] + nb[c];
        }
        *(unsigned*)&As[row][c16 * 4]     = packbf(y[0], y[1]);
        *(unsigned*)&As[row][c16 * 4 + 2] = packbf(y[2], y[3]);
    }
    __syncthreads();

    // --- stage 2: WMMA projections ---
    const int wave = tid >> 5, lane = tid & 31;
    const int arow = lane & 15, akb = (lane < 16) ? 0 : 16;
    const int Ml = (lane < 16) ? 0 : 8, Nl = lane & 15;

    for (int t = 0; t < 2; ++t) {
        const int nt = wave * 2 + t;       // 0..15
        v8f accv = {}; v8f accg = {};
        #pragma unroll
        for (int ks = 0; ks < 2; ++ks) {
            BF16Frag a, bv, bg;
            a.h[0] = *(const us8*)&As[arow][ks * 32 + akb];       // K = akb..akb+7
            a.h[1] = *(const us8*)&As[arow][ks * 32 + akb + 8];   // K = akb+8..akb+15
            const unsigned short* pm = WmB + (((nt * 2 + ks) * 32) + lane) * 16;
            const unsigned short* pg = WgB + (((nt * 2 + ks) * 32) + lane) * 16;
            bv.h[0] = *(const us8*)pm;  bv.h[1] = *(const us8*)(pm + 8);
            bg.h[0] = *(const us8*)pg;  bg.h[1] = *(const us8*)(pg + 8);
            accv = __builtin_amdgcn_wmma_f32_16x16x32_bf16(false, a.v, false, bv.v,
                                                           (short)0, accv, false, false);
            accg = __builtin_amdgcn_wmma_f32_16x16x32_bf16(false, a.v, false, bg.v,
                                                           (short)0, accg, false, false);
        }
        #pragma unroll
        for (int rr = 0; rr < 8; ++rr) {                           // C/D: M=rr or rr+8
            long r = r0 + rr + Ml;
            long col = (long)nt * 16 + Nl;
            vbuf[r * NDd + col] = f2bfbits(accv[rr]);
            gbuf[r * NDd + col] = f2bfbits(fastrcp(1.0f + __expf(-accg[rr])));
        }
    }
}

// ---------------------------------------------------------------------------
// Kernel 2: LayerNorm(z) + pair bias b[h,i,j] = zn@Wz. One wave32 per z row.
// ---------------------------------------------------------------------------
__global__ void k_ln_z_bias(const float* __restrict__ z,
                            const float* __restrict__ nw, const float* __restrict__ nb,
                            const float* __restrict__ Wz,
                            float* __restrict__ bbuf) {
    const int tid = threadIdx.x;
    const int lane = tid & 31, wave = tid >> 5;
    const long row = (long)blockIdx.x * 8 + wave;     // = i*512 + j
    const float* zr = z + row * CZd;
    float x[4]; float s = 0.f;
    #pragma unroll
    for (int q = 0; q < 4; ++q) { x[q] = zr[lane * 4 + q]; s += x[q]; }
    #pragma unroll
    for (int off = 16; off >= 1; off >>= 1) s += __shfl_xor(s, off, 32);
    const float mu = s * (1.0f / CZd);
    float vs = 0.f;
    #pragma unroll
    for (int q = 0; q < 4; ++q) { float d = x[q] - mu; vs += d * d; }
    #pragma unroll
    for (int off = 16; off >= 1; off >>= 1) vs += __shfl_xor(vs, off, 32);
    const float rs = rsqrtf(vs * (1.0f / CZd) + 1e-5f);
    float zn[4];
    #pragma unroll
    for (int q = 0; q < 4; ++q) {
        int c = lane * 4 + q;
        zn[q] = (x[q] - mu) * rs * nw[c] + nb[c];
    }
    #pragma unroll
    for (int h = 0; h < Hd; ++h) {
        float p = 0.f;
        #pragma unroll
        for (int q = 0; q < 4; ++q) p += zn[q] * Wz[(lane * 4 + q) * Hd + h];
        #pragma unroll
        for (int off = 16; off >= 1; off >>= 1) p += __shfl_xor(p, off, 32);
        if (lane == h) bbuf[(long)h * ROWS_Z + row] = p;
    }
}

// ---------------------------------------------------------------------------
// Kernel 3: softmax over j for each (h,i); write w bf16 row-major in j.
// ---------------------------------------------------------------------------
__global__ void k_softmax(const float* __restrict__ bbuf,
                          unsigned short* __restrict__ wbuf) {
    __shared__ float red[8];
    const int tid = threadIdx.x;
    const int h = blockIdx.x >> 9;
    const int i = blockIdx.x & 511;
    const float* brow = bbuf + (long)h * ROWS_Z + (long)i * Tdim;

    float v0 = brow[tid], v1 = brow[tid + 256];
    float mx = fmaxf(v0, v1);
    #pragma unroll
    for (int off = 16; off >= 1; off >>= 1) mx = fmaxf(mx, __shfl_xor(mx, off, 32));
    if ((tid & 31) == 0) red[tid >> 5] = mx;
    __syncthreads();
    mx = fmaxf(fmaxf(fmaxf(red[0], red[1]), fmaxf(red[2], red[3])),
               fmaxf(fmaxf(red[4], red[5]), fmaxf(red[6], red[7])));
    float e0 = __expf(v0 - mx), e1 = __expf(v1 - mx);
    float sm = e0 + e1;
    #pragma unroll
    for (int off = 16; off >= 1; off >>= 1) sm += __shfl_xor(sm, off, 32);
    __syncthreads();
    if ((tid & 31) == 0) red[tid >> 5] = sm;
    __syncthreads();
    sm = red[0] + red[1] + red[2] + red[3] + red[4] + red[5] + red[6] + red[7];
    const float inv = fastrcp(sm);        // feeds bf16, v_rcp_f32 is plenty
    unsigned short* wrow = wbuf + ((long)h * Tdim + i) * Tdim;
    wrow[tid]       = f2bfbits(e0 * inv);
    wrow[tid + 256] = f2bfbits(e1 * inv);
}

// ---------------------------------------------------------------------------
// Kernel 4: o[s,i,h,d] = sum_j w[h,i,j] * v[s,j,h,d]
//   Per h: GEMM M=512 (i), K=512 (j), N=16384 (n = s*32+d).
//   Block tile 32(M) x 128(N), 256 threads = 8 waves. Wave -> 1 M-tile x 2
//   N-tiles, 16 K-steps. V tile is staged in LDS in B-fragment-major order so
//   each fragment is two ds_load_b128. Per-head B (16 MB) is L2-resident.
// ---------------------------------------------------------------------------
__global__ void k_pair_avg(const unsigned short* __restrict__ wbuf,
                           const unsigned short* __restrict__ vbuf,
                           unsigned short* __restrict__ obuf) {
    __shared__ unsigned short Aw[32][32];          // w tile, row-major (i, j)
    __shared__ unsigned short BvF[8][32][16];      // v tile, fragment-major [nt][lane][e]
    const int tid = threadIdx.x;
    const int h  = blockIdx.z;
    const int i0 = blockIdx.y * 32;
    const int n0 = blockIdx.x * 128;
    const int wave = tid >> 5, lane = tid & 31;
    const int mi = wave >> 2, niBase = (wave & 3) * 2;
    const int arow = lane & 15, akb = (lane < 16) ? 0 : 16;

    v8f acc0 = {}, acc1 = {};
    for (int j0 = 0; j0 < Tdim; j0 += 32) {
        __syncthreads();
        // stage A: w[h, i0..i0+31, j0..j0+31] as 32-bit pairs (row-major)
        #pragma unroll
        for (int idx = tid; idx < 512; idx += 256) {
            int r = idx >> 4, cp = idx & 15;
            *(unsigned*)&Aw[r][cp * 2] =
                *(const unsigned*)&wbuf[((long)h * Tdim + (i0 + r)) * Tdim + j0 + cp * 2];
        }
        // stage B: v rows j0..j0+31, cols n0..n0+127, permuted to fragment order
        #pragma unroll
        for (int idx = tid; idx < 2048; idx += 256) {
            int r = idx >> 6, cp = idx & 63;          // k-row r, col pair cp
            int c = cp * 2;
            int n = n0 + c, s = n >> 5, d = n & 31;   // c even -> d,d+1 same s
            unsigned pr = *(const unsigned*)&vbuf[(((long)s * Tdim) + (j0 + r)) * NDd + h * Dd + d];
            int nt = c >> 4, ncol = c & 15;
            int lw = ncol + (r & 16);                 // B layout lane
            int e  = r & 15;
            BvF[nt][lw][e]     = (unsigned short)(pr & 0xFFFFu);
            BvF[nt][lw + 1][e] = (unsigned short)(pr >> 16);
        }
        // hint next A tile toward L2 while we compute
        if (j0 + 32 < Tdim)
            __builtin_prefetch(&wbuf[((long)h * Tdim + (i0 + (tid >> 3))) * Tdim + j0 + 32], 0, 1);
        __syncthreads();

        BF16Frag a, b0, b1;
        a.h[0]  = *(const us8*)&Aw[mi * 16 + arow][akb];
        a.h[1]  = *(const us8*)&Aw[mi * 16 + arow][akb + 8];
        b0.h[0] = *(const us8*)&BvF[niBase][lane][0];
        b0.h[1] = *(const us8*)&BvF[niBase][lane][8];
        b1.h[0] = *(const us8*)&BvF[niBase + 1][lane][0];
        b1.h[1] = *(const us8*)&BvF[niBase + 1][lane][8];
        acc0 = __builtin_amdgcn_wmma_f32_16x16x32_bf16(false, a.v, false, b0.v,
                                                       (short)0, acc0, false, false);
        acc1 = __builtin_amdgcn_wmma_f32_16x16x32_bf16(false, a.v, false, b1.v,
                                                       (short)0, acc1, false, false);
    }
    // store o as bf16 in [s, i, h, d] layout (contiguous for the final GEMM)
    const int Ml = (lane < 16) ? 0 : 8, Nl = lane & 15;
    #pragma unroll
    for (int t = 0; t < 2; ++t) {
        const v8f& acc = t ? acc1 : acc0;
        #pragma unroll
        for (int rr = 0; rr < 8; ++rr) {
            int i = i0 + mi * 16 + rr + Ml;
            int n = n0 + (niBase + t) * 16 + Nl;
            int s = n >> 5, d = n & 31;
            obuf[(((long)s * Tdim) + i) * NDd + h * Dd + d] = f2bfbits(acc[rr]);
        }
    }
}

// ---------------------------------------------------------------------------
// Kernel 5: out[r, c] = sum_hd (o[r,hd]*g[r,hd]) * Wo[hd, c]
//   16 rows/block, 128 threads = 4 waves (one 16-col N-tile each), 8 K-steps.
// ---------------------------------------------------------------------------
__global__ void k_gate_out(const unsigned short* __restrict__ obuf,
                           const unsigned short* __restrict__ gbuf,
                           const unsigned short* __restrict__ WoB,
                           float* __restrict__ out) {
    __shared__ unsigned short Xs[16][NDd];
    const int tid = threadIdx.x;
    const long r0 = (long)blockIdx.x * 16;
    #pragma unroll
    for (int idx = tid; idx < 2048; idx += 128) {   // 16*256 elems as 32-bit pairs
        int r = idx >> 7, cp = idx & 127;
        long a = (r0 + r) * NDd + cp * 2;
        unsigned o2 = *(const unsigned*)&obuf[a];
        unsigned g2 = *(const unsigned*)&gbuf[a];
        float x0 = bf2f(o2) * bf2f(g2);
        float x1 = bf2f(o2 >> 16) * bf2f(g2 >> 16);
        *(unsigned*)&Xs[r][cp * 2] = packbf(x0, x1);
    }
    __syncthreads();

    const int wave = tid >> 5, lane = tid & 31;     // wave = N-tile (CM=64 -> 4 tiles)
    const int arow = lane & 15, akb = (lane < 16) ? 0 : 16;
    v8f acc = {};
    #pragma unroll
    for (int ks = 0; ks < 8; ++ks) {
        BF16Frag a, b;
        a.h[0] = *(const us8*)&Xs[arow][ks * 32 + akb];
        a.h[1] = *(const us8*)&Xs[arow][ks * 32 + akb + 8];
        const unsigned short* pw = WoB + (((wave * 8 + ks) * 32) + lane) * 16;
        b.h[0] = *(const us8*)pw;  b.h[1] = *(const us8*)(pw + 8);
        acc = __builtin_amdgcn_wmma_f32_16x16x32_bf16(false, a.v, false, b.v,
                                                      (short)0, acc, false, false);
    }
    const int Ml = (lane < 16) ? 0 : 8, Nl = lane & 15;
    #pragma unroll
    for (int rr = 0; rr < 8; ++rr)
        out[(r0 + rr + Ml) * CMd + wave * 16 + Nl] = acc[rr];
}

// ---------------------------------------------------------------------------
// Host launcher
// ---------------------------------------------------------------------------
extern "C" void kernel_launch(void* const* d_in, const int* in_sizes, int n_in,
                              void* d_out, int out_size, void* d_ws, size_t ws_size,
                              hipStream_t stream) {
    const float* m   = (const float*)d_in[0];
    const float* z   = (const float*)d_in[1];
    const float* nmw = (const float*)d_in[2];
    const float* nmb = (const float*)d_in[3];
    const float* nzw = (const float*)d_in[4];
    const float* nzb = (const float*)d_in[5];
    const float* Wm  = (const float*)d_in[6];
    const float* Wg  = (const float*)d_in[7];
    const float* Wz  = (const float*)d_in[8];
    const float* Wo  = (const float*)d_in[9];
    float* out = (float*)d_out;

    // workspace carve-up (256B aligned)
    char* ws = (char*)d_ws;
    size_t off = 0;
    auto carve = [&](size_t bytes) {
        char* p = ws + off;
        off += (bytes + 255) & ~(size_t)255;
        return p;
    };
    unsigned short* wmB  = (unsigned short*)carve((size_t)CMd * NDd * 2);
    unsigned short* wgB  = (unsigned short*)carve((size_t)CMd * NDd * 2);
    unsigned short* woB  = (unsigned short*)carve((size_t)NDd * CMd * 2);
    unsigned short* vbuf = (unsigned short*)carve((size_t)ROWS_M * NDd * 2);   // 134 MB
    unsigned short* gbuf = (unsigned short*)carve((size_t)ROWS_M * NDd * 2);   // 134 MB
    unsigned short* obuf = (unsigned short*)carve((size_t)ROWS_M * NDd * 2);   // 134 MB
    float*          bbuf = (float*)carve((size_t)Hd * ROWS_Z * 4);             // 8.4 MB
    unsigned short* wbuf = (unsigned short*)carve((size_t)Hd * ROWS_Z * 2);    // 4.2 MB
    (void)ws_size;

    k_cvt_w<<<dim3((CMd * NDd + 255) / 256), dim3(256), 0, stream>>>(Wm, Wg, Wo, wmB, wgB, woB);

    k_ln_m_project<<<dim3(ROWS_M / 16), dim3(256), 0, stream>>>(
        m, nmw, nmb, wmB, wgB, vbuf, gbuf);

    k_ln_z_bias<<<dim3(ROWS_Z / 8), dim3(256), 0, stream>>>(
        z, nzw, nzb, Wz, bbuf);

    k_softmax<<<dim3(Hd * Tdim), dim3(256), 0, stream>>>(bbuf, wbuf);

    k_pair_avg<<<dim3((Sdim * Dd) / 128, Tdim / 32, Hd), dim3(256), 0, stream>>>(
        wbuf, vbuf, obuf);

    k_gate_out<<<dim3(ROWS_M / 16), dim3(128), 0, stream>>>(obuf, gbuf, woB, out);
}
